// UpCaps_68315749810635
// MI455X (gfx1250) — matrix operand order repaired and checked by reading, not compiled, not run on hardware
//
#include <hip/hip_runtime.h>
#include <hip/hip_bf16.h>

typedef __attribute__((ext_vector_type(16))) _Float16 v16h;
typedef __attribute__((ext_vector_type(8)))  float    v8f;

#define B_     8
#define CIN    64
#define Hh     64
#define Ww     64
#define ATOMS  16
#define COUT   64
#define Ktot   1600            // 5*5*64
#define KCH    50              // Ktot / 32
#define Mdim   1024            // COUT * ATOMS
#define Ndim   32768           // B * H * W
#define EPSc   1e-4f

// ---------------------------------------------------------------------------
// Workspace layout (bytes):
//   [0)              A_packed f16 : 1,638,400 halves = 3,276,800 B
//   [3,276,800)      xp f16       : 2,097,152 halves = 4,194,304 B
//   [7,471,104)      votes f32    : 33,554,432 f32   = 134,217,728 B
// total ~141.7 MB — votes + operands stay resident in the 192 MB L2.
// ---------------------------------------------------------------------------
#define AP_OFF   0
#define XP_OFF   3276800
#define VOTE_OFF 7471104

// -------- CDNA5 async global->LDS copy (ASYNCcnt path) ---------------------
// Inline asm: the builtin's pointer params use CUDA-LangAS-qualified int4*
// types that generic HIP pointers don't convert to; asm sidesteps that.
// vdst = per-lane LDS byte address (flat LDS-aperture ptr low 32 bits),
// vaddr = 64-bit global address pair, saddr = off (GV mode).
#if __has_builtin(__builtin_amdgcn_s_wait_asynccnt)
#define USE_ASYNC_LDS 1
__device__ __forceinline__ void async_cp16(void* lds_dst, const void* gsrc) {
  asm volatile("global_load_async_to_lds_b128 %0, %1, off"
               :
               : "v"((unsigned)(uintptr_t)lds_dst),
                 "v"((unsigned long long)(uintptr_t)gsrc)
               : "memory");
}
#define WAIT_ASYNC() __builtin_amdgcn_s_wait_asynccnt(0)
#else
#define USE_ASYNC_LDS 0
#define WAIT_ASYNC()
#endif

// Pack weights (KH,KW,CIN,ATOMS,COUT) f32 -> f16 in WMMA A-fragment order:
//   ap[((mt*KCH + kc)*32 + lane)*16 + i]
// A 16x32 f16 layout (contiguous k order used here, not the ISA table order): row M = lane%16;
//   K(lane,i) = i + (lane>=16 ? 16 : 0), per 32-wide K chunk.
__global__ void pack_w_kernel(const float* __restrict__ w, _Float16* __restrict__ ap) {
  int o    = blockIdx.x * 256 + threadIdx.x;     // 1,638,400 total
  int i    = o & 15;
  int lane = (o >> 4) & 31;
  int rem  = o >> 9;
  int kc   = rem % KCH;
  int mt   = rem / KCH;                          // == cout (m = mt*16 + atom)
  int c    = lane & 15;                          // atom
  int Kofs = i + ((lane >> 4) << 4);
  int Kt   = kc * 32 + Kofs;                     // global K index
  int khw  = Kt >> 6;                            // (kh*5 + kw)
  int cin  = Kt & 63;
  ap[o] = (_Float16)w[((khw * CIN + cin) * ATOMS + c) * COUT + mt];
}

// Pack x (B,CIN,H,W) f32 -> channels-last f16 xp[b][h][w][cin].
// LDS transpose: coalesced reads (consecutive w) and coalesced 32B writes.
__global__ __launch_bounds__(256)
void pack_x_kernel(const float* __restrict__ x, _Float16* __restrict__ xp) {
  __shared__ float tile[64][65];                 // +1 pad: conflict-free columns
  const int t  = threadIdx.x;
  const int bh = blockIdx.x;                     // b*64 + h, 512 blocks
  {
    int w = t & 63, c0 = t >> 6;                 // c0 in 0..3
#pragma unroll
    for (int j = 0; j < 16; ++j) {
      int cin = c0 + j * 4;
      tile[cin][w] = x[((size_t)(bh >> 6) * CIN + cin) * (Hh * Ww) +
                       (size_t)(bh & 63) * Ww + w];
    }
  }
  __syncthreads();
  {
    int w = t >> 2, c0 = (t & 3) * 16;
    v16h v;
#pragma unroll
    for (int q = 0; q < 16; ++q) v[q] = (_Float16)tile[c0 + q][w];
    *(v16h*)(xp + ((size_t)bh * Ww + w) * CIN + c0) = v;
  }
}

// Implicit-GEMM conv via WMMA, double-buffered LDS + async A copies.
// Grid: (Ndim/128, Mdim/256). 512 threads = 16 wave32s in a 4(M) x 4(N) grid.
// Each wave: 4x2 tiles of v_wmma_f32_16x16x32_f16 (64 acc VGPRs / lane).
// votes stored as [m][n], n = (h*64+w)*8 + b  (coalesced 64B row segments).
__global__ __launch_bounds__(512)
void conv_wmma_kernel(const _Float16* __restrict__ ap,
                      const _Float16* __restrict__ xp,
                      float* __restrict__ votes) {
  __shared__ v16h lds_a[2][16][32];  // 2 x 16 KB, pre-packed fragment order
  __shared__ v16h lds_b[2][8][32];   // 2 x  8 KB, fragment order

  const int tid  = threadIdx.x;
  const int lane = tid & 31;
  const int wv   = tid >> 5;       // 0..15
  const int wm   = wv >> 2;        // 0..3 (M direction)
  const int wn   = wv & 3;         // 0..3 (N direction)
  const int ng   = blockIdx.x;     // 0..255
  const int mg   = blockIdx.y;     // 0..3

  v8f acc[4][2] = {};

  // A staging constants (all 512 threads: one 32B slot each per chunk)
  const int amt = tid >> 5;
  const int al  = tid & 31;
  // B staging constants (threads 0..255 build the B fragments)
  const int bnt   = tid >> 5;                    // n-tile 0..7 (tid<256)
  const int nglob = ng * 128 + bnt * 16 + (lane & 15);
  const int pix   = nglob >> 3;                  // h*64 + w
  const int bb    = nglob & 7;                   // batch
  const int ph    = pix >> 6;
  const int pw    = pix & 63;
  const int cin_half = (lane >> 4) << 4;         // B frag: K = i + (lane>=16)*16

  // ---- prologue: stage chunk 0 into buffer 0 ----
  {
    const _Float16* g = ap + (size_t)(((mg * 16 + amt) * KCH + 0) * 32 + al) * 16;
    v16h* dst = &lds_a[0][amt][al];
#if USE_ASYNC_LDS
    async_cp16(dst, g);
    async_cp16((char*)dst + 16, (const char*)g + 16);
#else
    *dst = *(const v16h*)g;
#endif
    if (tid < 256) {
      int cin0 = cin_half;                       // kc=0: kh=kw=0, cin base 0
      int hh = ph - 2, wwp = pw - 2;
      v16h bv;
#pragma unroll
      for (int q = 0; q < 16; ++q) bv[q] = (_Float16)0.0f;
      if (hh >= 0 && wwp >= 0 && wwp < Ww)       // hh < 62 < Hh always
        bv = *(const v16h*)(xp + (size_t)(((bb * Hh + hh) * Ww + wwp) * CIN + cin0));
      lds_b[0][bnt][lane] = bv;
    }
    WAIT_ASYNC();
    __syncthreads();
  }

  for (int kc = 0; kc < KCH; ++kc) {
    const int cur = kc & 1, nxt = cur ^ 1;
    const bool have_next = (kc + 1 < KCH);

    // issue next chunk's loads first so they overlap the WMMAs below
    if (have_next) {
      const _Float16* g =
          ap + (size_t)(((mg * 16 + amt) * KCH + kc + 1) * 32 + al) * 16;
      v16h* dst = &lds_a[nxt][amt][al];
#if USE_ASYNC_LDS
      async_cp16(dst, g);
      async_cp16((char*)dst + 16, (const char*)g + 16);
#else
      *dst = *(const v16h*)g;
#endif
    }
    v16h bnext;
    if (have_next && tid < 256) {
      int kcn  = kc + 1;
      int khw  = kcn >> 1;
      int kh   = khw / 5, kw = khw % 5;
      int cin0 = ((kcn & 1) << 5) + cin_half;
      int hh   = ph + kh - 2;
      int wwp  = pw + kw - 2;
#pragma unroll
      for (int q = 0; q < 16; ++q) bnext[q] = (_Float16)0.0f;
      if (hh >= 0 && hh < Hh && wwp >= 0 && wwp < Ww)
        bnext = *(const v16h*)(xp + (size_t)(((bb * Hh + hh) * Ww + wwp) * CIN + cin0));
    }

    // compute on current buffer
    v16h bf0 = lds_b[cur][wn * 2 + 0][lane];
    v16h bf1 = lds_b[cur][wn * 2 + 1][lane];
#pragma unroll
    for (int mi = 0; mi < 4; ++mi) {
      v16h af = lds_a[cur][wm * 4 + mi][lane];
      acc[mi][0] = __builtin_amdgcn_wmma_f32_16x16x32_f16(
          false, af, false, bf0, (short)0, acc[mi][0], false, false);
      acc[mi][1] = __builtin_amdgcn_wmma_f32_16x16x32_f16(
          false, af, false, bf1, (short)0, acc[mi][1], false, false);
    }

    if (have_next && tid < 256) lds_b[nxt][bnt][lane] = bnext;
    WAIT_ASYNC();
    __syncthreads();
  }

  // Epilogue: C layout (ISA): element(r,lane) = (M = r + 8*(lane>=16), N = lane%16)
  const int n0 = ng * 128 + wn * 32 + (lane & 15);
  const int m0 = mg * 256 + wm * 64 + ((lane >> 4) << 3);
#pragma unroll
  for (int mi = 0; mi < 4; ++mi)
#pragma unroll
    for (int ni = 0; ni < 2; ++ni)
#pragma unroll
      for (int r = 0; r < 8; ++r)
        votes[(size_t)(m0 + mi * 16 + r) * Ndim + (n0 + ni * 16)] = acc[mi][ni][r];
}

// Dynamic routing, one workgroup per pixel. votes[m][pix*8+b], m = a*16 + c.
__global__ __launch_bounds__(256)
void routing_kernel(const float* __restrict__ votes, float* __restrict__ out) {
  __shared__ __align__(16) float Vs[Mdim * 8];   // 32 KB  [m][b]
  __shared__ float rs[64 * 8];                   // routing coeffs [a][b]
  __shared__ float ps[8 * 16];                   // preds [b][c]
  __shared__ float sqs[8];

  const int t = threadIdx.x;
  const int p = blockIdx.x;                      // pixel = h*64 + w

#pragma unroll
  for (int j = 0; j < 4; ++j) {
    int m = (t << 2) + j;
    const float4* src = (const float4*)(votes + (size_t)m * Ndim + p * 8);
    float4 lo = src[0], hi = src[1];
    float4* dst = (float4*)(Vs + m * 8);
    dst[0] = lo; dst[1] = hi;
  }
  __syncthreads();

  float l[8];
#pragma unroll
  for (int b = 0; b < 8; ++b) l[b] = 0.0f;

  for (int it = 0; it < 3; ++it) {
    // softmax over batch axis (faithful to the source's "dim 1")
    if (t < 64) {
      float e[8], s = 0.0f;
#pragma unroll
      for (int b = 0; b < 8; ++b) { e[b] = expf(l[b]); s += e[b]; }
      float inv = 1.0f / s;
#pragma unroll
      for (int b = 0; b < 8; ++b) rs[t * 8 + b] = e[b] * inv;
    }
    __syncthreads();
    // preds[b][c] = sum_a V[a,c,b] * r[a,b]
    if (t < 128) {
      int b = t >> 4, c = t & 15;
      float acc = 0.0f;
      for (int a = 0; a < 64; ++a)
        acc += Vs[(a * 16 + c) * 8 + b] * rs[a * 8 + b];
      ps[t] = acc;
    }
    __syncthreads();
    if (t < 8) {
      float s = EPSc;
#pragma unroll
      for (int c = 0; c < 16; ++c) { float v = ps[t * 16 + c]; s += v * v; }
      sqs[t] = s;
    }
    __syncthreads();
    if (t < 128) {
      float sq = sqs[t >> 4];
      float v  = ps[t];
      ps[t] = sq * v / ((1.0f + sq) * sqrtf(sq) + EPSc);
    }
    __syncthreads();
    // logits = r + sum_c V[a,c,b] * preds[b,c]   (skip on the last iteration)
    if (it < 2 && t < 64) {
#pragma unroll
      for (int b = 0; b < 8; ++b) {
        float dot = 0.0f;
#pragma unroll
        for (int c = 0; c < 16; ++c)
          dot += Vs[(t * 16 + c) * 8 + b] * ps[b * 16 + c];
        l[b] = rs[t * 8 + b] + dot;
      }
    }
    __syncthreads();
  }

  // out (B, ATOMS, H, W): index = (b*16+c)*4096 + p, with t = b*16+c
  if (t < 128) out[(size_t)t * 4096 + p] = ps[t];
}

extern "C" void kernel_launch(void* const* d_in, const int* in_sizes, int n_in,
                              void* d_out, int out_size, void* d_ws, size_t ws_size,
                              hipStream_t stream) {
  const float* x = (const float*)d_in[0];   // (8, 64, 64, 64)
  const float* w = (const float*)d_in[1];   // (5, 5, 64, 16, 64)
  float* out = (float*)d_out;               // (8, 16, 64, 64)

  char* ws = (char*)d_ws;                   // requires ~142 MB of scratch
  _Float16* ap    = (_Float16*)(ws + AP_OFF);
  _Float16* xp    = (_Float16*)(ws + XP_OFF);
  float*    votes = (float*)(ws + VOTE_OFF);

  pack_w_kernel<<<6400, 256, 0, stream>>>(w, ap);        // 1,638,400 elems
  pack_x_kernel<<<512, 256, 0, stream>>>(x, xp);         // one (b,h) row/block
  conv_wmma_kernel<<<dim3(Ndim / 128, Mdim / 256), 512, 0, stream>>>(ap, xp, votes);
  routing_kernel<<<4096, 256, 0, stream>>>(votes, out);
  (void)in_sizes; (void)n_in; (void)out_size; (void)ws_size;
}
